// TKL_3_42674795053909
// MI455X (gfx1250) — matrix operand order repaired and checked by reading, not compiled, not run on hardware
//
#include <hip/hip_runtime.h>
#include <cstdint>
#include <cstddef>

// ---------------------------------------------------------------------------
// Model constants (match reference)
// ---------------------------------------------------------------------------
#define BB      32
#define QQ      30
#define EMBD    128
#define NCHUNK  50
#define NSEQ    (BB*NCHUNK)     // 1600
#define EXTL    50
#define CHUNKL  40
#define OVL     5
#define DTOK    2000
#define NKER    11
#define NWIN0   988             // window 25
#define NWIN1   983             // window 35

typedef _Float16 h16 __attribute__((ext_vector_type(16)));
typedef _Float16 h8  __attribute__((ext_vector_type(8)));
typedef float    f8  __attribute__((ext_vector_type(8)));

// ---------------------------------------------------------------------------
// Wave helpers (wave32)
// ---------------------------------------------------------------------------
__device__ __forceinline__ float wsum(float v) {
#pragma unroll
  for (int o = 16; o; o >>= 1) v += __shfl_xor(v, o, 32);
  return v;
}
__device__ __forceinline__ float wmaxr(float v) {
#pragma unroll
  for (int o = 16; o; o >>= 1) v = fmaxf(v, __shfl_xor(v, o, 32));
  return v;
}

// ---------------------------------------------------------------------------
// WMMA fragment loaders, contiguous k order shared by A and B (wave32)
// A 16x32 f16: lane -> M=lane&15, K = (lane>>4)*16 .. +15
// B 32x16 f16: lane -> N=lane&15, K = (lane>>4)*16 .. +15  (from row-major W[N][K])
// D 16x16 f32: vgpr r -> M = (lane>>4)*8 + r, N = lane&15
// ---------------------------------------------------------------------------
__device__ __forceinline__ h16 frag_a_lds(const _Float16* A, int lda, int mt, int kt, int lane) {
  int m  = mt * 16 + (lane & 15);
  int hs = (lane >> 4) & 1;
  const _Float16* p = A + (size_t)m * lda + kt * 32 + hs * 16;
  h8 lo = *(const h8*)(p);
  h8 hi = *(const h8*)(p + 8);
  h16 a;
#pragma unroll
  for (int i = 0; i < 8; ++i) { a[i] = lo[i]; a[8 + i] = hi[i]; }
  return a;
}

__device__ __forceinline__ h16 frag_b_glb(const _Float16* W, int K, int nt, int kt, int lane) {
  int n = nt * 16 + (lane & 15);
  const _Float16* p = W + (size_t)n * K + kt * 32 + ((lane >> 4) & 1) * 16;
  h8 lo = *(const h8*)(p);
  h8 hi = *(const h8*)(p + 8);
  h16 b;
#pragma unroll
  for (int i = 0; i < 8; ++i) { b[i] = lo[i]; b[8 + i] = hi[i]; }
  return b;
}

__device__ __forceinline__ f8 wmma16(h16 a, h16 b, f8 c) {
  return __builtin_amdgcn_wmma_f32_16x16x32_f16(false, a, false, b, (short)0, c, false, false);
}

// ---------------------------------------------------------------------------
// Sinusoidal positional features (nts = 64, matches reference _pos_features)
// ---------------------------------------------------------------------------
__device__ __forceinline__ float posfeat(int pos, int d) {
  int k = (d < 64) ? d : d - 64;
  float t = (float)pos * __expf(-(float)k * (9.210340371976184f / 63.0f));
  return (d < 64) ? __sinf(t) : __cosf(t);
}

// ---------------------------------------------------------------------------
// small utility kernels
// ---------------------------------------------------------------------------
__global__ void cvt_kernel(const float* __restrict__ in, _Float16* __restrict__ out, int n) {
  int i = blockIdx.x * 256 + threadIdx.x;
  if (i < n) out[i] = (_Float16)in[i];
}

__global__ void qasm_kernel(const float* __restrict__ qe, const float* __restrict__ qm,
                            float* __restrict__ qin) {
  int i = blockIdx.x * 256 + threadIdx.x;
  if (i >= BB * QQ * EMBD) return;
  int d = i & 127, j = (i >> 7) % QQ, b = i / (QQ * EMBD);
  qin[i] = (qe[i] + posfeat(j, d)) * qm[b * QQ + j];
}

__global__ void dasm_kernel(const float* __restrict__ de, const float* __restrict__ dm,
                            float* __restrict__ din, float* __restrict__ cmask) {
  int i = blockIdx.x * 256 + threadIdx.x;   // < 1600*50*128
  int d = i & 127;
  int j = (i >> 7) % EXTL;
  int s = i / (EXTL * EMBD);
  int b = s / NCHUNK, c = s % NCHUNK;
  int p = c * CHUNKL - OVL + j;
  bool ok = (p >= 0) && (p < DTOK);
  float m = ok ? dm[b * DTOK + p] : 0.f;
  float e = ok ? de[((size_t)b * DTOK + p) * EMBD + d] : 0.f;
  din[i] = (e + posfeat(500 + j, d)) * m;
  if (d == 0) cmask[(size_t)s * EXTL + j] = m;
}

__global__ void packed_kernel(const float* __restrict__ cmask, float* __restrict__ packed) {
  int s = blockIdx.x * 256 + threadIdx.x;
  if (s >= NSEQ) return;
  float a = 0.f;
  for (int j = OVL; j < OVL + CHUNKL; ++j) a += cmask[(size_t)s * EXTL + j];
  packed[s] = (a != 0.f) ? 1.f : 0.f;
}

// ---------------------------------------------------------------------------
// LayerNorm over 128 dims, rows distributed to waves
// ---------------------------------------------------------------------------
__device__ __forceinline__ void lnorm128(float* xf, _Float16* xh, const float* g, const float* b,
                                         int SP, int wave, int lane) {
  for (int r = wave; r < SP; r += 8) {
    float x0 = xf[r * 128 + lane],      x1 = xf[r * 128 + lane + 32];
    float x2 = xf[r * 128 + lane + 64], x3 = xf[r * 128 + lane + 96];
    float mean = wsum(x0 + x1 + x2 + x3) * 0.0078125f;
    float d0 = x0 - mean, d1 = x1 - mean, d2 = x2 - mean, d3 = x3 - mean;
    float var = wsum(d0 * d0 + d1 * d1 + d2 * d2 + d3 * d3) * 0.0078125f;
    float rs = rsqrtf(var + 1e-5f);
    float o0 = d0 * rs * g[lane]      + b[lane];
    float o1 = d1 * rs * g[lane + 32] + b[lane + 32];
    float o2 = d2 * rs * g[lane + 64] + b[lane + 64];
    float o3 = d3 * rs * g[lane + 96] + b[lane + 96];
    xf[r * 128 + lane]      = o0; xh[r * 128 + lane]      = (_Float16)o0;
    xf[r * 128 + lane + 32] = o1; xh[r * 128 + lane + 32] = (_Float16)o1;
    xf[r * 128 + lane + 64] = o2; xh[r * 128 + lane + 64] = (_Float16)o2;
    xf[r * 128 + lane + 96] = o3; xh[r * 128 + lane + 96] = (_Float16)o3;
  }
}

// ---------------------------------------------------------------------------
// Fused 2-layer transformer encoder, one workgroup per sequence (256 thr / 8 waves)
// All GEMMs via v_wmma_f32_16x16x32_f16.
// MT is a template param (2: SP=32 query, 4: SP=64 doc chunks) so per-wave tile
// assignments, accumulator arrays and K-loops are fully compile-time unrolled.
// Big GEMMs register-block the A fragments (held across the nt sweep); Wo/FF2
// hold MT accumulators and stream one A fragment per K step.
// Dynamic LDS layout (bytes):
//   xf  f32 [64][128]  @      0  (32768)
//   sc  f32 [64][64]   @  32768  (16384)
//   xh  f16 [64][128]  @  49152  (16384)
//   ah  f16 [64][512]  @  65536  (65536)   (qkv ld=384, then FF1 out ld=512)
//   ath f16 [64][64]   @ 131072  ( 8192)
//   cxh f16 [64][128]  @ 139264  (16384)
//   msk f32 [64]       @ 155648  (  256)   total 155904
// ---------------------------------------------------------------------------
#define ENC_SMEM 155904

template <int MT>
__global__ void __launch_bounds__(256) encoder_kernel(
    const float* __restrict__ xin, const float* __restrict__ maskg,
    const _Float16* __restrict__ Wqkv, const float* __restrict__ bqkv,
    const _Float16* __restrict__ Wo,   const float* __restrict__ bo,
    const _Float16* __restrict__ Wf1,  const float* __restrict__ bf1,
    const _Float16* __restrict__ Wf2,  const float* __restrict__ bf2,
    const float* __restrict__ ln1g, const float* __restrict__ ln1b,
    const float* __restrict__ ln2g, const float* __restrict__ ln2b,
    const float* __restrict__ mixer,
    float* __restrict__ outn,
    int S, int out_start, int out_count) {
  extern __shared__ char smem[];
  float*    xf  = (float*)(smem);
  float*    sc  = (float*)(smem + 32768);
  _Float16* xh  = (_Float16*)(smem + 49152);
  _Float16* ah  = (_Float16*)(smem + 65536);
  _Float16* ath = (_Float16*)(smem + 131072);
  _Float16* cxh = (_Float16*)(smem + 139264);
  float*    msk = (float*)(smem + 155648);

  const int tid = threadIdx.x, wave = tid >> 5, lane = tid & 31;
  const int lm = lane & 15, hs = lane >> 4;
  const int seq = blockIdx.x;
  constexpr int SP = MT * 16;
  const int mtw = wave % MT;           // fixed M row-block per wave
  const int nt0 = wave / MT;
  constexpr int NTSTEP = 8 / MT;

  // load masked input (with pos already applied) into f32 + f16 LDS copies
  for (int i = tid; i < SP * 128; i += 256) {
    int m = i >> 7;
    float v = (m < S) ? xin[(size_t)seq * S * 128 + (size_t)m * 128 + (i & 127)] : 0.f;
    xf[i] = v; xh[i] = (_Float16)v;
  }
  if (tid < SP) msk[tid] = (tid < S) ? maskg[(size_t)seq * S + tid] : 0.f;
  __syncthreads();

  for (int l = 0; l < 2; ++l) {
    // -------- QKV projection: [SP x 128] x [128 x 384], fold 1/sqrt(dh) into q ----
    {
      const _Float16* W = Wqkv + (size_t)l * 384 * 128;
      const float*    b = bqkv + l * 384;
      for (int pf = tid; pf < 384; pf += 256)
        __builtin_prefetch(W + (size_t)pf * 128, 0, 1);   // global_prefetch_b8 -> L2
      h16 a0 = frag_a_lds(xh, 128, mtw, 0, lane);
      h16 a1 = frag_a_lds(xh, 128, mtw, 1, lane);
      h16 a2 = frag_a_lds(xh, 128, mtw, 2, lane);
      h16 a3 = frag_a_lds(xh, 128, mtw, 3, lane);
      for (int nt = nt0; nt < 24; nt += NTSTEP) {
        int n = nt * 16 + lm;
        float bv = b[n];
        f8 c;
#pragma unroll
        for (int r = 0; r < 8; ++r) c[r] = bv;
        c = wmma16(a0, frag_b_glb(W, 128, nt, 0, lane), c);
        c = wmma16(a1, frag_b_glb(W, 128, nt, 1, lane), c);
        c = wmma16(a2, frag_b_glb(W, 128, nt, 2, lane), c);
        c = wmma16(a3, frag_b_glb(W, 128, nt, 3, lane), c);
        float scl = (n < 128) ? 0.25f : 1.0f;
        int mb = mtw * 16 + hs * 8;
#pragma unroll
        for (int r = 0; r < 8; ++r) ah[(size_t)(mb + r) * 384 + n] = (_Float16)(c[r] * scl);
      }
    }
    __syncthreads();

    // -------- attention, head by head --------
    for (int h = 0; h < 8; ++h) {
      // scores = q @ k^T (K padded 16->32 with zeros)
      for (int t = wave; t < MT * MT; t += 8) {
        int mt = t % MT, nt = t / MT;
        int m = mt * 16 + lm;
        const _Float16* pq = ah + (size_t)m * 384 + h * 16;
        h16 a;
#pragma unroll
        for (int i = 0; i < 8; ++i) { a[i] = hs ? (_Float16)0.f : pq[i]; a[8 + i] = hs ? (_Float16)0.f : pq[8 + i]; }
        int ntok = nt * 16 + lm;
        h16 bk;
        if (hs == 0) {
          const _Float16* pk = ah + (size_t)ntok * 384 + 128 + h * 16;
#pragma unroll
          for (int i = 0; i < 16; ++i) bk[i] = pk[i];
        } else {
#pragma unroll
          for (int i = 0; i < 16; ++i) bk[i] = (_Float16)0.f;
        }
        f8 c;
#pragma unroll
        for (int r = 0; r < 8; ++r) c[r] = 0.f;
        c = wmma16(a, bk, c);
        int mb = mt * 16 + hs * 8;
#pragma unroll
        for (int r = 0; r < 8; ++r) sc[(mb + r) * 64 + nt * 16 + lm] = c[r];
      }
      __syncthreads();
      // masked softmax per row
      for (int r = wave; r < SP; r += 8) {
        float v0 = sc[r * 64 + lane] - (1.f - msk[lane]) * 1e9f;
        float v1 = (SP == 64) ? (sc[r * 64 + 32 + lane] - (1.f - msk[32 + lane]) * 1e9f) : -3.0e38f;
        float mx = wmaxr(fmaxf(v0, v1));
        float e0 = __expf(v0 - mx);
        float e1 = (SP == 64) ? __expf(v1 - mx) : 0.f;
        float inv = 1.f / wsum(e0 + e1);
        ath[r * 64 + lane] = (_Float16)(e0 * inv);
        if (SP == 64) ath[r * 64 + 32 + lane] = (_Float16)(e1 * inv);
      }
      __syncthreads();
      // ctx = attn @ v
      for (int mt = wave; mt < MT; mt += 8) {
        f8 c;
#pragma unroll
        for (int r = 0; r < 8; ++r) c[r] = 0.f;
#pragma unroll
        for (int kt = 0; kt < SP / 32; ++kt) {
          h16 a = frag_a_lds(ath, 64, mt, kt, lane);
          h16 bv;
#pragma unroll
          for (int i = 0; i < 16; ++i)
            bv[i] = ah[(size_t)(kt * 32 + hs * 16 + i) * 384 + 256 + h * 16 + lm];
          c = wmma16(a, bv, c);
        }
        int mb = mt * 16 + hs * 8;
#pragma unroll
        for (int r = 0; r < 8; ++r) cxh[(mb + r) * 128 + h * 16 + lm] = (_Float16)c[r];
      }
      __syncthreads();
    }

    // -------- Wo projection + residual (MT accumulators, stream A per K step) ----
    {
      const _Float16* W = Wo + (size_t)l * 128 * 128;
      const float*    b = bo + l * 128;
      for (int pf = tid; pf < 128; pf += 256)
        __builtin_prefetch(W + (size_t)pf * 128, 0, 1);
      f8 acc[MT];
#pragma unroll
      for (int j = 0; j < MT; ++j) {
        float bv = b[(nt0 + j * NTSTEP) * 16 + lm];
#pragma unroll
        for (int r = 0; r < 8; ++r) acc[j][r] = bv;
      }
#pragma unroll
      for (int kt = 0; kt < 4; ++kt) {
        h16 a = frag_a_lds(cxh, 128, mtw, kt, lane);
#pragma unroll
        for (int j = 0; j < MT; ++j)
          acc[j] = wmma16(a, frag_b_glb(W, 128, nt0 + j * NTSTEP, kt, lane), acc[j]);
      }
      int mb = mtw * 16 + hs * 8;
#pragma unroll
      for (int j = 0; j < MT; ++j) {
        int n = (nt0 + j * NTSTEP) * 16 + lm;
#pragma unroll
        for (int r = 0; r < 8; ++r) xf[(mb + r) * 128 + n] += acc[j][r];
      }
    }
    __syncthreads();
    lnorm128(xf, xh, ln1g + l * 128, ln1b + l * 128, SP, wave, lane);
    __syncthreads();

    // -------- FF1 + relu (A fragments held in registers across nt sweep) --------
    {
      const _Float16* W = Wf1 + (size_t)l * 512 * 128;
      const float*    b = bf1 + l * 512;
      for (int pf = tid; pf < 512; pf += 256)
        __builtin_prefetch(W + (size_t)pf * 128, 0, 1);
      h16 a0 = frag_a_lds(xh, 128, mtw, 0, lane);
      h16 a1 = frag_a_lds(xh, 128, mtw, 1, lane);
      h16 a2 = frag_a_lds(xh, 128, mtw, 2, lane);
      h16 a3 = frag_a_lds(xh, 128, mtw, 3, lane);
      for (int nt = nt0; nt < 32; nt += NTSTEP) {
        int n = nt * 16 + lm;
        float bv = b[n];
        f8 c;
#pragma unroll
        for (int r = 0; r < 8; ++r) c[r] = bv;
        c = wmma16(a0, frag_b_glb(W, 128, nt, 0, lane), c);
        c = wmma16(a1, frag_b_glb(W, 128, nt, 1, lane), c);
        c = wmma16(a2, frag_b_glb(W, 128, nt, 2, lane), c);
        c = wmma16(a3, frag_b_glb(W, 128, nt, 3, lane), c);
        int mb = mtw * 16 + hs * 8;
#pragma unroll
        for (int r = 0; r < 8; ++r) ah[(size_t)(mb + r) * 512 + n] = (_Float16)fmaxf(c[r], 0.f);
      }
    }
    __syncthreads();

    // -------- FF2 + residual (MT accumulators, K=512 streamed) --------
    {
      const _Float16* W = Wf2 + (size_t)l * 128 * 512;
      const float*    b = bf2 + l * 128;
      for (int pf = tid; pf < 128; pf += 256)
        __builtin_prefetch(W + (size_t)pf * 512, 0, 1);
      f8 acc[MT];
#pragma unroll
      for (int j = 0; j < MT; ++j) {
        float bv = b[(nt0 + j * NTSTEP) * 16 + lm];
#pragma unroll
        for (int r = 0; r < 8; ++r) acc[j][r] = bv;
      }
#pragma unroll
      for (int kt = 0; kt < 16; ++kt) {
        h16 a = frag_a_lds(ah, 512, mtw, kt, lane);
#pragma unroll
        for (int j = 0; j < MT; ++j)
          acc[j] = wmma16(a, frag_b_glb(W, 512, nt0 + j * NTSTEP, kt, lane), acc[j]);
      }
      int mb = mtw * 16 + hs * 8;
#pragma unroll
      for (int j = 0; j < MT; ++j) {
        int n = (nt0 + j * NTSTEP) * 16 + lm;
#pragma unroll
        for (int r = 0; r < 8; ++r) xf[(mb + r) * 128 + n] += acc[j][r];
      }
    }
    __syncthreads();
    lnorm128(xf, xh, ln2g + l * 128, ln2b + l * 128, SP, wave, lane);
    __syncthreads();
  }

  // -------- mix with input, mask, L2-normalize, write out rows --------
  float mixv = mixer[0];
  for (int r = wave; r < out_count; r += 8) {
    int m = out_start + r;
    float y[4]; float ss = 0.f;
#pragma unroll
    for (int j = 0; j < 4; ++j) {
      int cidx = lane + 32 * j;
      float x0 = xin[(size_t)seq * S * 128 + (size_t)m * 128 + cidx];
      float v = (mixv * x0 + (1.f - mixv) * xf[m * 128 + cidx]) * msk[m];
      y[j] = v; ss += v * v;
    }
    float inv = 1.f / (sqrtf(wsum(ss)) + 1e-13f);
#pragma unroll
    for (int j = 0; j < 4; ++j)
      outn[((size_t)seq * out_count + r) * 128 + lane + 32 * j] = y[j] * inv;
  }
}

// ---------------------------------------------------------------------------
// Fused cosine + RBF kernels + prefix-scan along 2000 doc tokens.
// One block per (b,q); 12 waves: waves 0..10 scan kernel k, wave 11 scans nz.
// csum layout: [b*30+q][2001][12]
// ---------------------------------------------------------------------------
__global__ void __launch_bounds__(384) cosscan_kernel(
    const float* __restrict__ qn, const float* __restrict__ dn,
    const float* __restrict__ cmask, const float* __restrict__ packed,
    float* __restrict__ csum) {
  __shared__ __align__(16) float qv[128];
  __shared__ float cosb[DTOK];
  __shared__ float mb[DTOK];
  int tid = threadIdx.x;
  int b = blockIdx.x / QQ, q = blockIdx.x % QQ;
  for (int i = tid; i < 128; i += 384) qv[i] = qn[((size_t)b * QQ + q) * 128 + i];
  __syncthreads();
  for (int t = tid; t < DTOK; t += 384) {
    int s = b * NCHUNK + t / CHUNKL, j = t % CHUNKL;
    const float4* dp = (const float4*)(dn + ((size_t)s * CHUNKL + j) * 128);
    const float4* qp = (const float4*)qv;
    float acc = 0.f;
#pragma unroll
    for (int u = 0; u < 32; ++u) {
      float4 a = dp[u], c = qp[u];
      acc += a.x * c.x + a.y * c.y + a.z * c.z + a.w * c.w;
    }
    cosb[t] = acc;
    mb[t] = cmask[(size_t)s * EXTL + OVL + j] * packed[s];
  }
  __syncthreads();
  int wave = tid >> 5, lane = tid & 31;
  if (wave < 12) {
    int w = wave;
    float mu  = (w == 0) ? 1.0f : (1.1f - 0.2f * (float)w);
    float i2s = (w == 0) ? 5.0e5f : 50.0f;   // 1/(2*sigma^2)
    size_t base = ((size_t)(b * QQ + q)) * 2001 * 12 + w;
    if (lane == 0) csum[base] = 0.f;
    float carry = 0.f;
    for (int t0 = 0; t0 < DTOK; t0 += 32) {
      int t = t0 + lane;
      float v = 0.f;
      if (t < DTOK) {
        if (w == 11) v = (mb[t] != 0.f) ? 1.f : 0.f;
        else { float d = cosb[t] - mu; v = __expf(-d * d * i2s) * mb[t]; }
      }
#pragma unroll
      for (int o = 1; o < 32; o <<= 1) {
        float u = __shfl_up(v, o, 32);
        if (lane >= o) v += u;
      }
      v += carry;
      if (t < DTOK) csum[base + (size_t)(t + 1) * 12] = v;
      carry = __shfl(v, 31, 32);
    }
  }
}

// ---------------------------------------------------------------------------
// Sliding-window saturation scoring. grid (32, 2), 256 threads.
// ---------------------------------------------------------------------------
__global__ void __launch_bounds__(256) score_kernel(
    const float* __restrict__ csum, const float* __restrict__ qidf,
    const float* __restrict__ qmask,
    const float* __restrict__ sW1, const float* __restrict__ sb1,
    const float* __restrict__ sW2, const float* __restrict__ sb2,
    const float* __restrict__ sW3, const float* __restrict__ sb3,
    const float* __restrict__ dW, float* __restrict__ score) {
  int b = blockIdx.x, win = blockIdx.y;
  int w = win ? 35 : 25;
  int n_win = win ? NWIN1 : NWIN0;
  float w10 = sW1[0], w11 = sW1[1], c1 = sb1[0];
  float w20 = sW2[0], w21 = sW2[1], c2 = sb2[0];
  float w30 = sW3[0], w31 = sW3[1], c3 = sb3[0];
  for (int i = threadIdx.x; i < n_win; i += 256) {
    float pk[NKER];
#pragma unroll
    for (int k = 0; k < NKER; ++k) pk[k] = 0.f;
    for (int q = 0; q < QQ; ++q) {
      size_t rb = ((size_t)(b * QQ + q) * 2001 + 2 * i) * 12;
      size_t re = ((size_t)(b * QQ + q) * 2001 + 2 * i + w) * 12;
      float len = csum[re + 11] - csum[rb + 11];
      float qmv = qmask[b * QQ + q];
      if (len > 0.f && qmv != 0.f) {
        float f0 = qidf[b * QQ + q];
        float s1 = w10 * f0 + w11 * len + c1;
        float s2 = 1.f / (w20 * f0 + w21 * len + c2);
        float s3 = w30 * f0 + w31 * len + c3;
#pragma unroll
        for (int k = 0; k < NKER; ++k) {
          float p = csum[re + k] - csum[rb + k];
          p = fmaxf(p, 1e-10f);
          pk[k] += s1 * __powf(p, s2) - s3;
        }
      }
    }
    float s = 0.f;
#pragma unroll
    for (int k = 0; k < NKER; ++k) s += pk[k] * dW[k];
    if (s == 0.f) s = -9900.f;
    score[((size_t)(b * 2 + win)) * 1024 + i] = s;
  }
}

// ---------------------------------------------------------------------------
// top-3 region pooling (exact replication of reference) + final blend
// ---------------------------------------------------------------------------
__global__ void topk_kernel(const float* __restrict__ score, const float* __restrict__ csc,
                            const float* __restrict__ sws, float* __restrict__ out) {
  __shared__ float ybuf[64];
  int t = threadIdx.x;
  if (t < 64) {
    int b = t >> 1, win = t & 1;
    float w = win ? 35.f : 25.f;
    int n_win = win ? NWIN1 : NWIN0;
    const float* s = score + (size_t)(b * 2 + win) * 1024;
    int best = 0; float bv = s[0];
    for (int i = 1; i < n_win; ++i)
      if (s[i] > bv) { bv = s[i]; best = i; }
    float mpr[91];
    for (int r = 0; r < 91; ++r) mpr[r] = s[r];
    float y = 0.f;
    for (int c = 0; c < 3; ++c) {
      float val = s[best];
      if (val <= -9900.f) val = 0.f;
      y += val * csc[c];
      for (int r = 0; r < 91; ++r)
        if (fabsf((float)(r - best)) < w * 0.5f) mpr[r] = -10001.f - (float)c;
      best = 0; bv = mpr[0];
      for (int r = 1; r < 91; ++r)
        if (mpr[r] > bv) { bv = mpr[r]; best = r; }
    }
    ybuf[t] = y;
  }
  __syncthreads();
  if (t < 32) {
    float s = sws[0];
    float wa = (35.f - s) * 0.1f, wb = (s - 25.f) * 0.1f;
    out[t] = wa * ybuf[t * 2] + wb * ybuf[t * 2 + 1];
  }
}

// ---------------------------------------------------------------------------
// host launcher
// ---------------------------------------------------------------------------
extern "C" void kernel_launch(void* const* d_in, const int* in_sizes, int n_in,
                              void* d_out, int out_size, void* d_ws, size_t ws_size,
                              hipStream_t stream) {
  (void)in_sizes; (void)n_in; (void)out_size; (void)ws_size;
  const float* qe    = (const float*)d_in[0];
  const float* de    = (const float*)d_in[1];
  const float* qmask = (const float*)d_in[2];
  const float* dmask = (const float*)d_in[3];
  const float* qidf  = (const float*)d_in[4];
  // d_in[5] = document_idfs (unused by reference)
  const float* mixer = (const float*)d_in[6];
  const float* Wqkv  = (const float*)d_in[7];
  const float* bqkv  = (const float*)d_in[8];
  const float* Wo    = (const float*)d_in[9];
  const float* bo    = (const float*)d_in[10];
  const float* Wf1   = (const float*)d_in[11];
  const float* bf1   = (const float*)d_in[12];
  const float* Wf2   = (const float*)d_in[13];
  const float* bf2   = (const float*)d_in[14];
  const float* ln1g  = (const float*)d_in[15];
  const float* ln1b  = (const float*)d_in[16];
  const float* ln2g  = (const float*)d_in[17];
  const float* ln2b  = (const float*)d_in[18];
  const float* sW1   = (const float*)d_in[19];
  const float* sb1   = (const float*)d_in[20];
  const float* sW2   = (const float*)d_in[21];
  const float* sb2   = (const float*)d_in[22];
  const float* sW3   = (const float*)d_in[23];
  const float* sb3   = (const float*)d_in[24];
  const float* dW    = (const float*)d_in[25];
  const float* csc   = (const float*)d_in[26];
  const float* sws   = (const float*)d_in[27];

  char* ws = (char*)d_ws;
  size_t off = 0;
  auto alloc = [&](size_t bytes) -> void* {
    void* p = ws + off;
    off += (bytes + 255) & ~(size_t)255;
    return p;
  };
  const int nWqkv = 2 * 384 * 128, nWo = 2 * 128 * 128;
  const int nWf1 = 2 * 512 * 128, nWf2 = 2 * 128 * 512;
  _Float16* hWqkv = (_Float16*)alloc((size_t)nWqkv * 2);
  _Float16* hWo   = (_Float16*)alloc((size_t)nWo * 2);
  _Float16* hWf1  = (_Float16*)alloc((size_t)nWf1 * 2);
  _Float16* hWf2  = (_Float16*)alloc((size_t)nWf2 * 2);
  float* qin    = (float*)alloc((size_t)BB * QQ * EMBD * 4);
  float* din    = (float*)alloc((size_t)NSEQ * EXTL * EMBD * 4);
  float* cmaskw = (float*)alloc((size_t)NSEQ * EXTL * 4);
  float* packedw= (float*)alloc((size_t)NSEQ * 4);
  float* qn     = (float*)alloc((size_t)BB * QQ * EMBD * 4);
  float* dn     = (float*)alloc((size_t)NSEQ * CHUNKL * EMBD * 4);
  float* csumw  = (float*)alloc((size_t)BB * QQ * 2001 * 12 * 4);
  float* scorew = (float*)alloc((size_t)BB * 2 * 1024 * 4);

  // weight precision conversion
  cvt_kernel<<<(nWqkv + 255) / 256, 256, 0, stream>>>(Wqkv, hWqkv, nWqkv);
  cvt_kernel<<<(nWo + 255) / 256, 256, 0, stream>>>(Wo, hWo, nWo);
  cvt_kernel<<<(nWf1 + 255) / 256, 256, 0, stream>>>(Wf1, hWf1, nWf1);
  cvt_kernel<<<(nWf2 + 255) / 256, 256, 0, stream>>>(Wf2, hWf2, nWf2);

  // sequence assembly (pos-enc + mask, chunking)
  qasm_kernel<<<(BB * QQ * EMBD + 255) / 256, 256, 0, stream>>>(qe, qmask, qin);
  dasm_kernel<<<(NSEQ * EXTL * EMBD) / 256, 256, 0, stream>>>(de, dmask, din, cmaskw);
  packed_kernel<<<(NSEQ + 255) / 256, 256, 0, stream>>>(cmaskw, packedw);

  // WMMA transformer encoders
  hipFuncSetAttribute(reinterpret_cast<const void*>(&encoder_kernel<2>),
                      hipFuncAttributeMaxDynamicSharedMemorySize, ENC_SMEM);
  hipFuncSetAttribute(reinterpret_cast<const void*>(&encoder_kernel<4>),
                      hipFuncAttributeMaxDynamicSharedMemorySize, ENC_SMEM);
  encoder_kernel<2><<<BB, 256, ENC_SMEM, stream>>>(
      qin, qmask, hWqkv, bqkv, hWo, bo, hWf1, bf1, hWf2, bf2,
      ln1g, ln1b, ln2g, ln2b, mixer, qn, QQ, 0, QQ);
  encoder_kernel<4><<<NSEQ, 256, ENC_SMEM, stream>>>(
      din, cmaskw, hWqkv, bqkv, hWo, bo, hWf1, bf1, hWf2, bf2,
      ln1g, ln1b, ln2g, ln2b, mixer, dn, EXTL, OVL, CHUNKL);

  // kernel-pooling cumulative sums
  cosscan_kernel<<<BB * QQ, 384, 0, stream>>>(qn, dn, cmaskw, packedw, csumw);

  // window scoring + pooling + blend
  dim3 sg(BB, 2);
  score_kernel<<<sg, 256, 0, stream>>>(csumw, qidf, qmask, sW1, sb1, sW2, sb2, sW3, sb3,
                                       dW, scorew);
  topk_kernel<<<1, 64, 0, stream>>>(scorew, csc, sws, (float*)d_out);
}
